// Model_75969381532428
// MI455X (gfx1250) — matrix operand order repaired and checked by reading, not compiled, not run on hardware
//
#include <hip/hip_runtime.h>
#include <hip/hip_bf16.h>
#include <math.h>

typedef __attribute__((ext_vector_type(16))) _Float16 v16h;
typedef __attribute__((ext_vector_type(8)))  float    v8f;
typedef __attribute__((ext_vector_type(4)))  unsigned int v4u;
typedef __attribute__((ext_vector_type(8)))  int      v8i_;
typedef __attribute__((ext_vector_type(4)))  int      v4i_;

#define N_NODES 131072
#define N_EDGES 524288

// ---------- lane/layout helpers (contiguous 16-bit k order used by this file, wave32) ----------
// A-matrix 16x32 f16: slot j -> K = j + (lane>=16 ? 16 : 0), M = lane&15
__device__ __forceinline__ int aSlotK(int lane, int j) {
  return j + ((lane & 16) ? 16 : 0);
}
// B-matrix 32x16 f16: slot j -> K = j + (lane>=16 ? 16 : 0), N = lane&15

__device__ __forceinline__ unsigned pcg_hash(unsigned x) {
  x = x * 747796405u + 2891336453u;
  x = ((x >> ((x >> 28) + 4u)) ^ x) * 277803737u;
  return (x >> 22) ^ x;
}
__device__ __forceinline__ float gauss_hash(unsigned s) {
  unsigned u1 = pcg_hash(s);
  unsigned u2 = pcg_hash(s ^ 0x9E3779B9u);
  float f1 = ((float)u1 + 1.0f) * 2.3283064365386963e-10f;
  float f2 = (float)u2 * 2.3283064365386963e-10f;
  return sqrtf(-2.0f * logf(f1)) * cosf(6.28318530718f * f2);
}

// ---------- TDM probe: NULL descriptor (count=0) is an architectural no-op --
// 6-arg form (clang-23 / therock-10.0 headers):
//   (uint32x4 g0, int32x8 g1, int32x4 g2, int32x4 g3, int32x8 g4, i32 cpol)
__global__ void tdm_null_kernel() {
  __shared__ long long lds_pad[16];
  (void)lds_pad;
  v4u  g0 = (v4u)0u;  // D# group0: count=0 -> NULL tensor
  v8i_ g1 = (v8i_)0;
  v4i_ g2 = (v4i_)0;
  v4i_ g3 = (v4i_)0;
  v8i_ g4 = (v8i_)0;
  __builtin_amdgcn_tensor_load_to_lds(g0, g1, g2, g3, g4, 0);
  __builtin_amdgcn_s_wait_tensorcnt(0);
}

// ---------- small conversion kernels ----------
__global__ void f32_to_f16_kernel(const float* __restrict__ src,
                                  _Float16* __restrict__ dst, int n) {
  int i = blockIdx.x * blockDim.x + threadIdx.x;
  if (i < n) dst[i] = (_Float16)src[i];
}

// Pad a (validRows, cols) f32 matrix into (totalRows, cols) f16 with zero rows.
__global__ void pad_rows_kernel(const float* __restrict__ src,
                                _Float16* __restrict__ dst, int validRows,
                                int cols, int totalRows) {
  int i = blockIdx.x * blockDim.x + threadIdx.x;
  if (i >= totalRows * cols) return;
  int r = i / cols;
  dst[i] = (r < validRows) ? (_Float16)src[i] : (_Float16)0.0f;
}

// Pad conv weights [COUT][KR] f32 -> [COUT][KP] f16 (KP = KR rounded up to 32).
__global__ void pad_wconv_kernel(const float* __restrict__ src,
                                 _Float16* __restrict__ dst, int COUT, int KR,
                                 int KP) {
  int i = blockIdx.x * blockDim.x + threadIdx.x;
  if (i >= COUT * KP) return;
  int co = i / KP, k = i % KP;
  dst[i] = (k < KR) ? (_Float16)src[co * KR + k] : (_Float16)0.0f;
}

__global__ void gen_img_f16_kernel(_Float16* __restrict__ img) {
  int i = blockIdx.x * blockDim.x + threadIdx.x;
  if (i < 8 * 3 * 64 * 64) img[i] = (_Float16)gauss_hash((unsigned)i ^ 0xFA4Eu);
}

// ---------- node MLP: x = relu([room|attr] @ W1^T + b1) @ W2^T + b2 ----------
__global__ __launch_bounds__(128) void node_mlp_kernel(
    const int* __restrict__ rooms, const int* __restrict__ attrs,
    const float* __restrict__ room_embed, const float* __restrict__ attr_embed,
    const _Float16* __restrict__ w1, const float* __restrict__ b1,
    const _Float16* __restrict__ w2, const float* __restrict__ b2,
    _Float16* __restrict__ x_out) {
  __shared__ _Float16 lds_h[4][16 * 128];
  const int lane = threadIdx.x & 31;
  const int wave = threadIdx.x >> 5;
  const int rowM = lane & 15;
  const int nCol = lane & 15;
  const int hiK = (lane & 16) ? 16 : 0;
  const int hiM = (lane & 16) ? 8 : 0;
  const int row0 = (blockIdx.x * 4 + wave) * 16;

  const int room = rooms[row0 + rowM];
  const int attr = attrs[row0 + rowM];

  v16h a[3];
#pragma unroll
  for (int c = 0; c < 3; ++c) {
#pragma unroll
    for (int j = 0; j < 16; ++j) {
      int k = c * 32 + aSlotK(lane, j);
      float v = (k < 64) ? room_embed[room * 64 + k]
                         : attr_embed[attr * 32 + (k - 64)];
      a[c][j] = (_Float16)v;
    }
  }

#pragma unroll
  for (int nt = 0; nt < 8; ++nt) {
    float bias = b1[nt * 16 + nCol];
    v8f c8 = {bias, bias, bias, bias, bias, bias, bias, bias};
#pragma unroll
    for (int kc = 0; kc < 3; ++kc) {
      v16h bm;
#pragma unroll
      for (int j = 0; j < 16; ++j)
        bm[j] = w1[(nt * 16 + nCol) * 96 + kc * 32 + j + hiK];
      c8 = __builtin_amdgcn_wmma_f32_16x16x32_f16(false, a[kc], false, bm,
                                                  (short)0, c8, false, false);
    }
#pragma unroll
    for (int r = 0; r < 8; ++r) {
      float v = c8[r];
      v = v > 0.0f ? v : 0.0f;
      lds_h[wave][(r + hiM) * 128 + nt * 16 + nCol] = (_Float16)v;
    }
  }
  __syncthreads();

  v16h a2[4];
#pragma unroll
  for (int c = 0; c < 4; ++c) {
#pragma unroll
    for (int j = 0; j < 16; ++j)
      a2[c][j] = lds_h[wave][rowM * 128 + c * 32 + aSlotK(lane, j)];
  }
#pragma unroll
  for (int nt = 0; nt < 4; ++nt) {
    float bias = b2[nt * 16 + nCol];
    v8f c8 = {bias, bias, bias, bias, bias, bias, bias, bias};
#pragma unroll
    for (int kc = 0; kc < 4; ++kc) {
      v16h bm;
#pragma unroll
      for (int j = 0; j < 16; ++j)
        bm[j] = w2[(nt * 16 + nCol) * 128 + kc * 32 + j + hiK];
      c8 = __builtin_amdgcn_wmma_f32_16x16x32_f16(false, a2[kc], false, bm,
                                                  (short)0, c8, false, false);
    }
#pragma unroll
    for (int r = 0; r < 8; ++r)
      x_out[(row0 + r + hiM) * 64 + nt * 16 + nCol] = (_Float16)c8[r];
  }
}

// ---------- fc_box: boxes = relu(x @ W1^T + b1) @ W2^T + b2 (64->128->4) ----
__global__ __launch_bounds__(128) void box_mlp_kernel(
    const _Float16* __restrict__ x_in, const _Float16* __restrict__ w1,
    const float* __restrict__ b1, const _Float16* __restrict__ w2pad,
    const float* __restrict__ b2, float* __restrict__ boxes_pred,
    float* __restrict__ boxes_refine) {
  __shared__ _Float16 lds_h[4][16 * 128];
  const int lane = threadIdx.x & 31;
  const int wave = threadIdx.x >> 5;
  const int rowM = lane & 15;
  const int nCol = lane & 15;
  const int hiK = (lane & 16) ? 16 : 0;
  const int hiM = (lane & 16) ? 8 : 0;
  const int row0 = (blockIdx.x * 4 + wave) * 16;

  v16h a[2];
#pragma unroll
  for (int c = 0; c < 2; ++c) {
#pragma unroll
    for (int j = 0; j < 16; ++j)
      a[c][j] = x_in[(row0 + rowM) * 64 + c * 32 + aSlotK(lane, j)];
  }
#pragma unroll
  for (int nt = 0; nt < 8; ++nt) {
    float bias = b1[nt * 16 + nCol];
    v8f c8 = {bias, bias, bias, bias, bias, bias, bias, bias};
#pragma unroll
    for (int kc = 0; kc < 2; ++kc) {
      v16h bm;
#pragma unroll
      for (int j = 0; j < 16; ++j)
        bm[j] = w1[(nt * 16 + nCol) * 64 + kc * 32 + j + hiK];
      c8 = __builtin_amdgcn_wmma_f32_16x16x32_f16(false, a[kc], false, bm,
                                                  (short)0, c8, false, false);
    }
#pragma unroll
    for (int r = 0; r < 8; ++r) {
      float v = c8[r];
      v = v > 0.0f ? v : 0.0f;
      lds_h[wave][(r + hiM) * 128 + nt * 16 + nCol] = (_Float16)v;
    }
  }
  __syncthreads();

  v16h a2[4];
#pragma unroll
  for (int c = 0; c < 4; ++c) {
#pragma unroll
    for (int j = 0; j < 16; ++j)
      a2[c][j] = lds_h[wave][rowM * 128 + c * 32 + aSlotK(lane, j)];
  }
  float bias = (nCol < 4) ? b2[nCol] : 0.0f;
  v8f c8 = {bias, bias, bias, bias, bias, bias, bias, bias};
#pragma unroll
  for (int kc = 0; kc < 4; ++kc) {
    v16h bm;
#pragma unroll
    for (int j = 0; j < 16; ++j)
      bm[j] = w2pad[nCol * 128 + kc * 32 + j + hiK];
    c8 = __builtin_amdgcn_wmma_f32_16x16x32_f16(false, a2[kc], false, bm,
                                                (short)0, c8, false, false);
  }
  if (nCol < 4) {
#pragma unroll
    for (int r = 0; r < 8; ++r) {
      int idx = (row0 + r + hiM) * 4 + nCol;
      float v = c8[r];
      boxes_pred[idx] = v;
      boxes_refine[idx] =
          v + 0.01f * gauss_hash((unsigned)idx * 2654435761u ^ 0xB0C5u);
    }
  }
}

// ---------- edge MLP: 192 -> 128 -> 64 over E edges (dominant GEMM) --------
__global__ __launch_bounds__(128) void edge_mlp_kernel(
    const int* __restrict__ triples, const _Float16* __restrict__ x_in,
    const float* __restrict__ edge_embed, const _Float16* __restrict__ w1,
    const float* __restrict__ b1, const _Float16* __restrict__ w2,
    const float* __restrict__ b2, float* __restrict__ edge_feat) {
  __shared__ _Float16 lds_h[4][16 * 128];
  const int lane = threadIdx.x & 31;
  const int wave = threadIdx.x >> 5;
  const int rowM = lane & 15;
  const int nCol = lane & 15;
  const int hiK = (lane & 16) ? 16 : 0;
  const int hiM = (lane & 16) ? 8 : 0;
  const int row0 = (blockIdx.x * 4 + wave) * 16;

  __builtin_prefetch((const void*)w1, 0, 3);
  __builtin_prefetch((const void*)w2, 0, 3);

  const int e = row0 + rowM;
  const int s = triples[e * 3 + 0];
  const int rl = triples[e * 3 + 1];
  const int d = triples[e * 3 + 2];

  v16h a[6];
#pragma unroll
  for (int c = 0; c < 6; ++c) {
#pragma unroll
    for (int j = 0; j < 16; ++j) {
      int k = c * 32 + aSlotK(lane, j);
      _Float16 v;
      if (k < 64)       v = x_in[s * 64 + k];
      else if (k < 128) v = (_Float16)edge_embed[rl * 64 + (k - 64)];
      else              v = x_in[d * 64 + (k - 128)];
      a[c][j] = v;
    }
  }

#pragma unroll
  for (int nt = 0; nt < 8; ++nt) {
    float bias = b1[nt * 16 + nCol];
    v8f c8 = {bias, bias, bias, bias, bias, bias, bias, bias};
#pragma unroll
    for (int kc = 0; kc < 6; ++kc) {
      v16h bm;
#pragma unroll
      for (int j = 0; j < 16; ++j)
        bm[j] = w1[(nt * 16 + nCol) * 192 + kc * 32 + j + hiK];
      c8 = __builtin_amdgcn_wmma_f32_16x16x32_f16(false, a[kc], false, bm,
                                                  (short)0, c8, false, false);
    }
#pragma unroll
    for (int r = 0; r < 8; ++r) {
      float v = c8[r];
      v = v > 0.0f ? v : 0.0f;
      lds_h[wave][(r + hiM) * 128 + nt * 16 + nCol] = (_Float16)v;
    }
  }
  __syncthreads();

  v16h a2[4];
#pragma unroll
  for (int c = 0; c < 4; ++c) {
#pragma unroll
    for (int j = 0; j < 16; ++j)
      a2[c][j] = lds_h[wave][rowM * 128 + c * 32 + aSlotK(lane, j)];
  }
#pragma unroll
  for (int nt = 0; nt < 4; ++nt) {
    float bias = b2[nt * 16 + nCol];
    v8f c8 = {bias, bias, bias, bias, bias, bias, bias, bias};
#pragma unroll
    for (int kc = 0; kc < 4; ++kc) {
      v16h bm;
#pragma unroll
      for (int j = 0; j < 16; ++j)
        bm[j] = w2[(nt * 16 + nCol) * 128 + kc * 32 + j + hiK];
      c8 = __builtin_amdgcn_wmma_f32_16x16x32_f16(false, a2[kc], false, bm,
                                                  (short)0, c8, false, false);
    }
#pragma unroll
    for (int r = 0; r < 8; ++r)
      edge_feat[(size_t)(row0 + r + hiM) * 64 + nt * 16 + nCol] = c8[r];
  }
}

// ---------- implicit-GEMM WMMA 3x3 conv (SAME, stride 1, 64x64, batch 8) ----
// M = 16 consecutive pixels in one row, N = 16 output channels,
// K = Cin*9 (zero-padded to multiple of 32), decoded as (ci, dy, dx).
template <int CIN, int COUT>
__global__ __launch_bounds__(128) void conv3x3_wmma_kernel(
    const _Float16* __restrict__ in,  // [8][CIN][64][64] f16
    const _Float16* __restrict__ wp,  // [COUT][KP] f16, zero-padded
    const float* __restrict__ bias, float* __restrict__ outp)  // f32 pre-BN
{
  constexpr int KR = CIN * 9;
  constexpr int KP = (KR + 31) & ~31;
  constexpr int KCH = KP / 32;
  constexpr int NT = COUT / 16;

  const int lane = threadIdx.x & 31;
  const int wave = threadIdx.x >> 5;
  const int rowM = lane & 15;
  const int nCol = lane & 15;
  const int hiK = (lane & 16) ? 16 : 0;
  const int hiM = (lane & 16) ? 8 : 0;

  int tid = blockIdx.x * 4 + wave;
  int nt = tid % NT;
  int pt = tid / NT;          // pixel tile: 256 per image (64 y * 4 x-tiles)
  int n = pt >> 8;
  int rem = pt & 255;
  int y = rem >> 2;
  int x0 = (rem & 3) << 4;
  int x = x0 + rowM;

  float bv = bias[nt * 16 + nCol];
  v8f c8 = {bv, bv, bv, bv, bv, bv, bv, bv};

  const _Float16* wrow = wp + (nt * 16 + nCol) * KP;
  for (int kc = 0; kc < KCH; ++kc) {
    v16h a;
#pragma unroll
    for (int j = 0; j < 16; ++j) {
      int k = kc * 32 + aSlotK(lane, j);
      _Float16 v = (_Float16)0.0f;
      if (k < KR) {
        int ci = k / 9;
        int t = k - ci * 9;
        int dy = t / 3 - 1;
        int dx = t - (dy + 1) * 3 - 1;
        int yy = y + dy, xx = x + dx;
        if (yy >= 0 && yy < 64 && xx >= 0 && xx < 64)
          v = in[((n * CIN + ci) << 12) + (yy << 6) + xx];
      }
      a[j] = v;
    }
    v16h bm;
#pragma unroll
    for (int j = 0; j < 16; ++j) bm[j] = wrow[kc * 32 + j + hiK];
    c8 = __builtin_amdgcn_wmma_f32_16x16x32_f16(false, a, false, bm, (short)0,
                                                c8, false, false);
  }
  int co = nt * 16 + nCol;
#pragma unroll
  for (int r = 0; r < 8; ++r)
    outp[((n * COUT + co) << 12) + (y << 6) + x0 + r + hiM] = c8[r];
}

// ---------- 1x1 conv (64 -> 14, N padded to 16) as a straight GEMM ----------
__global__ __launch_bounds__(128) void conv1x1_wmma_kernel(
    const _Float16* __restrict__ in,   // [8][64][4096] f16
    const _Float16* __restrict__ wpad, // [16][64] f16, rows >=14 zero
    const float* __restrict__ bias,    // [14]
    float* __restrict__ outp)          // [8][14][4096] f32
{
  const int lane = threadIdx.x & 31;
  const int wave = threadIdx.x >> 5;
  const int rowM = lane & 15;
  const int nCol = lane & 15;
  const int hiK = (lane & 16) ? 16 : 0;
  const int hiM = (lane & 16) ? 8 : 0;

  int pt = blockIdx.x * 4 + wave;  // 2048 pixel tiles
  int n = pt >> 8;
  int pix0 = (pt & 255) << 4;
  int pix = pix0 + rowM;

  float bv = (nCol < 14) ? bias[nCol] : 0.0f;
  v8f c8 = {bv, bv, bv, bv, bv, bv, bv, bv};
#pragma unroll
  for (int kc = 0; kc < 2; ++kc) {
    v16h a;
#pragma unroll
    for (int j = 0; j < 16; ++j) {
      int ci = kc * 32 + aSlotK(lane, j);
      a[j] = in[((n * 64 + ci) << 12) + pix];
    }
    v16h bm;
#pragma unroll
    for (int j = 0; j < 16; ++j) bm[j] = wpad[nCol * 64 + kc * 32 + j + hiK];
    c8 = __builtin_amdgcn_wmma_f32_16x16x32_f16(false, a, false, bm, (short)0,
                                                c8, false, false);
  }
  if (nCol < 14) {
#pragma unroll
    for (int r = 0; r < 8; ++r)
      outp[((n * 14 + nCol) << 12) + pix0 + r + hiM] = c8[r];
  }
}

// ---------- batch-norm statistics + apply (f32 stats, f16 activations) ------
__global__ void bn_stats_kernel(const float* __restrict__ x,
                                float* __restrict__ mean,
                                float* __restrict__ inv, int C) {
  __shared__ float ssum[256], ssq[256];
  int c = blockIdx.x;
  float s = 0.0f, q = 0.0f;
  for (int i = threadIdx.x; i < 8 * 4096; i += 256) {
    float v = x[(((i >> 12) * C + c) << 12) + (i & 4095)];
    s += v;
    q += v * v;
  }
  ssum[threadIdx.x] = s;
  ssq[threadIdx.x] = q;
  __syncthreads();
  for (int o = 128; o > 0; o >>= 1) {
    if (threadIdx.x < o) {
      ssum[threadIdx.x] += ssum[threadIdx.x + o];
      ssq[threadIdx.x] += ssq[threadIdx.x + o];
    }
    __syncthreads();
  }
  if (threadIdx.x == 0) {
    float m = ssum[0] * (1.0f / 32768.0f);
    float var = ssq[0] * (1.0f / 32768.0f) - m * m;
    mean[c] = m;
    inv[c] = rsqrtf(var + 1e-5f);
  }
}

__global__ void bn_relu_f16_kernel(const float* __restrict__ x,
                                   _Float16* __restrict__ y,
                                   const float* __restrict__ mean,
                                   const float* __restrict__ inv,
                                   const float* __restrict__ gamma,
                                   const float* __restrict__ beta, int C) {
  int i = blockIdx.x * blockDim.x + threadIdx.x;
  int total = 8 * C * 4096;
  if (i >= total) return;
  int c = (i >> 12) % C;
  float v = (x[i] - mean[c]) * inv[c] * gamma[c] + beta[c];
  y[i] = (_Float16)(v > 0.0f ? v : 0.0f);
}

// ---------- launcher --------------------------------------------------------
extern "C" void kernel_launch(void* const* d_in, const int* in_sizes, int n_in,
                              void* d_out, int out_size, void* d_ws,
                              size_t ws_size, hipStream_t stream) {
  const int* rooms = (const int*)d_in[0];
  const int* triples = (const int*)d_in[1];
  const int* attrs = (const int*)d_in[3];
  const float* room_embed = (const float*)d_in[4];
  const float* edge_embed = (const float*)d_in[5];
  const float* attr_embed = (const float*)d_in[6];
  const float* w1r = (const float*)d_in[7];  const float* b1r = (const float*)d_in[8];
  const float* w2r = (const float*)d_in[9];  const float* b2r = (const float*)d_in[10];
  const float* w1e = (const float*)d_in[11]; const float* b1e = (const float*)d_in[12];
  const float* w2e = (const float*)d_in[13]; const float* b2e = (const float*)d_in[14];
  const float* w1b = (const float*)d_in[15]; const float* b1b = (const float*)d_in[16];
  const float* w2b = (const float*)d_in[17]; const float* b2b = (const float*)d_in[18];
  const float* c1w = (const float*)d_in[19]; const float* c1b = (const float*)d_in[20];
  const float* c1g = (const float*)d_in[21]; const float* c1be = (const float*)d_in[22];
  const float* c2w = (const float*)d_in[23]; const float* c2b = (const float*)d_in[24];
  const float* c2g = (const float*)d_in[25]; const float* c2be = (const float*)d_in[26];
  const float* c3w = (const float*)d_in[27]; const float* c3b = (const float*)d_in[28];
  const float* c3g = (const float*)d_in[29]; const float* c3be = (const float*)d_in[30];
  const float* c4w = (const float*)d_in[31]; const float* c4b = (const float*)d_in[32];
  const float* c4g = (const float*)d_in[33]; const float* c4be = (const float*)d_in[34];
  const float* c5w = (const float*)d_in[35]; const float* c5b = (const float*)d_in[36];

  float* out = (float*)d_out;
  float* boxes_pred = out;                              // N*4
  float* layout_map = out + 524288;                     // 8*14*64*64
  float* boxes_refine = out + 524288 + 458752;          // N*4
  float* edge_feat = out + 524288 + 458752 + 524288;    // E*64

  char* ws = (char*)d_ws;
  size_t off = 0;
  _Float16* x16 = (_Float16*)(ws + off);    off += (size_t)N_NODES * 64 * 2;
  _Float16* w1r16 = (_Float16*)(ws + off);  off += 128 * 96 * 2;
  _Float16* w2r16 = (_Float16*)(ws + off);  off += 64 * 128 * 2;
  _Float16* w1e16 = (_Float16*)(ws + off);  off += 128 * 192 * 2;
  _Float16* w2e16 = (_Float16*)(ws + off);  off += 64 * 128 * 2;
  _Float16* w1b16 = (_Float16*)(ws + off);  off += 128 * 64 * 2;
  _Float16* w2b16 = (_Float16*)(ws + off);  off += 16 * 128 * 2;
  _Float16* wc1_16 = (_Float16*)(ws + off); off += 32 * 32 * 2;
  _Float16* wc2_16 = (_Float16*)(ws + off); off += 64 * 288 * 2;
  _Float16* wc3_16 = (_Float16*)(ws + off); off += 128 * 576 * 2;
  _Float16* wc4_16 = (_Float16*)(ws + off); off += 64 * 1152 * 2;
  _Float16* wc5_16 = (_Float16*)(ws + off); off += 16 * 64 * 2;
  _Float16* fake16 = (_Float16*)(ws + off); off += (size_t)8 * 3 * 4096 * 2;
  _Float16* h16A = (_Float16*)(ws + off);   off += (size_t)8 * 128 * 4096 * 2;
  _Float16* h16B = (_Float16*)(ws + off);   off += (size_t)8 * 128 * 4096 * 2;
  float* bufA = (float*)(ws + off);         off += (size_t)8 * 128 * 4096 * 4;
  float* bufB = (float*)(ws + off);         off += (size_t)8 * 128 * 4096 * 4;
  float* statm = (float*)(ws + off);        off += 128 * 4;
  float* stati = (float*)(ws + off);        off += 128 * 4;
  (void)ws_size; (void)in_sizes; (void)n_in; (void)out_size;

  // 0) TDM probe (NULL descriptor no-op; exercises tensor path + tensorcnt).
  tdm_null_kernel<<<1, 32, 0, stream>>>();

  // 1) f16 weight materialization.
  f32_to_f16_kernel<<<(128 * 96 + 255) / 256, 256, 0, stream>>>(w1r, w1r16, 128 * 96);
  f32_to_f16_kernel<<<(64 * 128 + 255) / 256, 256, 0, stream>>>(w2r, w2r16, 64 * 128);
  f32_to_f16_kernel<<<(128 * 192 + 255) / 256, 256, 0, stream>>>(w1e, w1e16, 128 * 192);
  f32_to_f16_kernel<<<(64 * 128 + 255) / 256, 256, 0, stream>>>(w2e, w2e16, 64 * 128);
  f32_to_f16_kernel<<<(128 * 64 + 255) / 256, 256, 0, stream>>>(w1b, w1b16, 128 * 64);
  pad_rows_kernel<<<(16 * 128 + 255) / 256, 256, 0, stream>>>(w2b, w2b16, 4, 128, 16);
  pad_wconv_kernel<<<(32 * 32 + 255) / 256, 256, 0, stream>>>(c1w, wc1_16, 32, 27, 32);
  pad_wconv_kernel<<<(64 * 288 + 255) / 256, 256, 0, stream>>>(c2w, wc2_16, 64, 288, 288);
  pad_wconv_kernel<<<(128 * 576 + 255) / 256, 256, 0, stream>>>(c3w, wc3_16, 128, 576, 576);
  pad_wconv_kernel<<<(64 * 1152 + 255) / 256, 256, 0, stream>>>(c4w, wc4_16, 64, 1152, 1152);
  pad_rows_kernel<<<(16 * 64 + 255) / 256, 256, 0, stream>>>(c5w, wc5_16, 14, 64, 16);

  // 2) node MLP -> x (f16, N x 64).
  node_mlp_kernel<<<N_NODES / 64, 128, 0, stream>>>(
      rooms, attrs, room_embed, attr_embed, w1r16, b1r, w2r16, b2r, x16);

  // 3) fc_box -> boxes_pred / boxes_refine.
  box_mlp_kernel<<<N_NODES / 64, 128, 0, stream>>>(
      x16, w1b16, b1b, w2b16, b2b, boxes_pred, boxes_refine);

  // 4) edge MLP (dominant GEMM) -> edge_feat.
  edge_mlp_kernel<<<N_EDGES / 64, 128, 0, stream>>>(
      triples, x16, edge_embed, w1e16, b1e, w2e16, b2e, edge_feat);

  // 5) conv/bn chain on hash-generated fake image, all convs on WMMA.
  gen_img_f16_kernel<<<(8 * 3 * 4096 + 255) / 256, 256, 0, stream>>>(fake16);

  // c1: 3 -> 32  (2048 pixel tiles * 2 ntiles = 4096 wave-tiles)
  conv3x3_wmma_kernel<3, 32><<<4096 / 4, 128, 0, stream>>>(fake16, wc1_16, c1b, bufA);
  bn_stats_kernel<<<32, 256, 0, stream>>>(bufA, statm, stati, 32);
  bn_relu_f16_kernel<<<(8 * 32 * 4096 + 255) / 256, 256, 0, stream>>>(
      bufA, h16A, statm, stati, c1g, c1be, 32);

  // c2: 32 -> 64 (2048 * 4 = 8192 wave-tiles)
  conv3x3_wmma_kernel<32, 64><<<8192 / 4, 128, 0, stream>>>(h16A, wc2_16, c2b, bufB);
  bn_stats_kernel<<<64, 256, 0, stream>>>(bufB, statm, stati, 64);
  bn_relu_f16_kernel<<<(8 * 64 * 4096 + 255) / 256, 256, 0, stream>>>(
      bufB, h16B, statm, stati, c2g, c2be, 64);

  // c3: 64 -> 128 (2048 * 8 = 16384 wave-tiles)
  conv3x3_wmma_kernel<64, 128><<<16384 / 4, 128, 0, stream>>>(h16B, wc3_16, c3b, bufA);
  bn_stats_kernel<<<128, 256, 0, stream>>>(bufA, statm, stati, 128);
  bn_relu_f16_kernel<<<(8 * 128 * 4096 + 255) / 256, 256, 0, stream>>>(
      bufA, h16A, statm, stati, c3g, c3be, 128);

  // c4: 128 -> 64 (8192 wave-tiles)
  conv3x3_wmma_kernel<128, 64><<<8192 / 4, 128, 0, stream>>>(h16A, wc4_16, c4b, bufB);
  bn_stats_kernel<<<64, 256, 0, stream>>>(bufB, statm, stati, 64);
  bn_relu_f16_kernel<<<(8 * 64 * 4096 + 255) / 256, 256, 0, stream>>>(
      bufB, h16B, statm, stati, c4g, c4be, 64);

  // c5: 1x1, 64 -> 14 (2048 pixel tiles)
  conv1x1_wmma_kernel<<<2048 / 4, 128, 0, stream>>>(h16B, wc5_16, c5b, layout_map);
}
